// AttentionOnSteps_63488206570043
// MI455X (gfx1250) — matrix operand order repaired and checked by reading, not compiled, not run on hardware
//
#include <hip/hip_runtime.h>
#include <hip/hip_bf16.h>

// ---------------------------------------------------------------------------
// Attention pipeline for MI455X (gfx1250).
//   K1: Q/K/V = x @ W[z]            (fp32->bf16 convert in VGPRs, WMMA bf16)
//   K2: S = Q @ K^T * 1/sqrt(D)     (TDM double-buffered 128x64 tiles + WMMA)
//   K3: row softmax over S in place
//   K4: out = P @ V                 (TDM double-buffered P tile + WMMA)
// Workspace: Qb@0, Kb@16M, Vb@32M, S@48M  (80 MB, intermediates live in L2)
// ---------------------------------------------------------------------------

typedef __attribute__((ext_vector_type(16))) __bf16 v16bf;
typedef __attribute__((ext_vector_type(8)))  __bf16 v8bf;
typedef __attribute__((ext_vector_type(8)))  float  v8f;
typedef __attribute__((ext_vector_type(4)))  unsigned int u32x4;
typedef __attribute__((ext_vector_type(8)))  unsigned int u32x8;

#define Dq   1024
#define Sq   2048
#define Bq   4
#define Mtot (Bq * Sq)      // 8192
#define LDP32 40            // LDS row stride for 32-elem rows (64B + 16B pad)
#define LDP64 72            // LDS row stride for 64-elem rows (128B + 16B pad)

static __device__ inline unsigned short f2bf_u(float f) {
    union { float f; unsigned u; } c; c.f = f;
    unsigned u = c.u;
    u += 0x7FFFu + ((u >> 16) & 1u);          // round-to-nearest-even
    return (unsigned short)(u >> 16);
}
static __device__ inline __bf16 f2bf(float f) {
    unsigned short s = f2bf_u(f);
    __bf16 r; __builtin_memcpy(&r, &s, 2); return r;
}
static __device__ inline float bf2f(__bf16 b) {
    unsigned short s; __builtin_memcpy(&s, &b, 2);
    union { unsigned u; float f; } c; c.u = ((unsigned)s) << 16;
    return c.f;
}

// --------------------------- TDM (Tensor Data Mover) -----------------------
// 2D bf16 tile load Global->LDS: tile_d1 rows x tile_d0 elems, memory row
// stride stride_d0 elems. LDS rows padded +16B via D# pad fields:
//   pad_iv = 3 -> pad every 16 DWORDs (64B row,  BK=32)
//   pad_iv = 4 -> pad every 32 DWORDs (128B row, BK=64)
//   pad_am = 3 -> insert 4 DWORDs (16B)
static __device__ inline void tdm_load_tile_bf16(const void* gptr, unsigned lds_byte_off,
                                                 unsigned tensor_d0, unsigned tensor_d1,
                                                 unsigned tile_d0, unsigned tile_d1,
                                                 unsigned stride_d0, unsigned pad_iv) {
    unsigned long long ga = (unsigned long long)gptr;
    u32x4 g0;
    g0[0] = 1u;                                       // count=1 (valid user D#)
    g0[1] = lds_byte_off;                             // lds_addr
    g0[2] = (unsigned)(ga & 0xFFFFFFFFu);             // global_addr[31:0]
    g0[3] = (unsigned)((ga >> 32) & 0x01FFFFFFu)      // global_addr[56:32]
          | (2u << 30);                               // type = 2 (image)
    u32x8 g1;
    g1[0] = (1u << 16)                                // data_size = 2 bytes
          | (1u << 20)                                // pad_enable
          | (pad_iv << 22)                            // pad_interval
          | (3u << 25);                               // pad_amount: 4 DWORDs (16B)
    g1[1] = (tensor_d0 & 0xFFFFu) << 16;              // tensor_dim0[15:0]
    g1[2] = (tensor_d0 >> 16) | ((tensor_d1 & 0xFFFFu) << 16);
    g1[3] = (tensor_d1 >> 16) | (tile_d0 << 16);      // tile_dim0
    g1[4] = tile_d1;                                  // tile_dim1 (tile_dim2 = 0)
    g1[5] = stride_d0;                                // tensor_dim0_stride[31:0]
    g1[6] = 0u;
    g1[7] = 0u;
    asm volatile("tensor_load_to_lds %0, %1" :: "s"(g0), "s"(g1) : "memory");
}
static __device__ inline unsigned lds_off(const void* p) {
    return (unsigned)(unsigned long long)p;           // low 32 bits = LDS byte offset
}

// --------------------------- WMMA fragments --------------------------------
// A fragment (16x32 bf16, rows in LDS row-major):
// lane: m = lane&15, elems [koff..koff+7, 16+koff..16+koff+7], koff=(lane>>4)*8
static __device__ inline v16bf load_frag_a(const __bf16* lds, int row0, int stride) {
    const int lane = threadIdx.x & 31;
    const int r    = row0 + (lane & 15);
    const int koff = (lane >> 4) << 3;
    const __bf16* p = lds + r * stride + koff;
    v8bf lo = *(const v8bf*)(p);
    v8bf hi = *(const v8bf*)(p + 16);
    v16bf f;
#pragma unroll
    for (int i = 0; i < 8; ++i) { f[i] = lo[i]; f[i + 8] = hi[i]; }
    return f;
}
// B fragment (32x16 bf16) from K-major tile tileT[n][k]:
// lane: n = lane&15, elems [kbase..kbase+7, 16+kbase..16+kbase+7], kbase=(lane>>4)*8
static __device__ inline v16bf load_frag_b(const __bf16* ldsT, int col0, int stride) {
    const int lane  = threadIdx.x & 31;
    const int n     = col0 + (lane & 15);
    const int kbase = (lane >> 4) << 3;
    const __bf16* p = ldsT + n * stride + kbase;
    v8bf lo = *(const v8bf*)(p);
    v8bf hi = *(const v8bf*)(p + 16);
    v16bf f;
#pragma unroll
    for (int i = 0; i < 8; ++i) { f[i] = lo[i]; f[i + 8] = hi[i]; }
    return f;
}
static __device__ inline v8f wmma_bf16(v16bf a, v16bf b, v8f c) {
    return __builtin_amdgcn_wmma_f32_16x16x32_bf16(
        false, a, false, b, (short)0, c, false, false);
}

// ---------------------------------------------------------------------------
// K1: QKV projection. grid(8, 64, 3), block 256. BK=32, VGPR convert path.
// ---------------------------------------------------------------------------
__global__ __launch_bounds__(256) void qkv_gemm(const float* __restrict__ x,
                                                const float* __restrict__ w,
                                                __bf16* __restrict__ qkv) {
    __shared__ __align__(16) __bf16 As[128 * LDP32];
    __shared__ __align__(16) __bf16 Bt[128 * LDP32];

    const int tid  = threadIdx.x;
    const int wave = tid >> 5;
    const int lane = tid & 31;
    const int gm0  = blockIdx.y * 128;
    const int gn0  = blockIdx.x * 128;
    const float* wz = w + (size_t)blockIdx.z * Dq * Dq;
    __bf16* out = qkv + (size_t)blockIdx.z * Mtot * Dq;

    const int wm = (wave >> 2) * 64;
    const int wn = (wave & 3) * 32;

    v8f acc[4][2] = {};

    for (int k0 = 0; k0 < Dq; k0 += 32) {
#pragma unroll
        for (int p = 0; p < 4; ++p) {       // x tile 128x32 fp32->bf16
            int idx = tid + p * 256;
            int row = idx >> 3;
            int c4  = (idx & 7) << 2;
            float4 v = *(const float4*)(x + (size_t)(gm0 + row) * Dq + k0 + c4);
            ushort4 h;
            h.x = f2bf_u(v.x); h.y = f2bf_u(v.y); h.z = f2bf_u(v.z); h.w = f2bf_u(v.w);
            *(ushort4*)(&As[row * LDP32 + c4]) = h;
        }
#pragma unroll
        for (int p = 0; p < 4; ++p) {       // W tile 32x128 transposed -> Bt[n][k]
            int e  = tid & 127;
            int g  = (tid >> 7) + p * 2;
            int kk = g << 2;
            const float* wp = wz + (size_t)(k0 + kk) * Dq + gn0 + e;
            ushort4 h;
            h.x = f2bf_u(wp[0 * Dq]); h.y = f2bf_u(wp[1 * Dq]);
            h.z = f2bf_u(wp[2 * Dq]); h.w = f2bf_u(wp[3 * Dq]);
            *(ushort4*)(&Bt[e * LDP32 + kk]) = h;
        }
        if (k0 + 32 < Dq) {   // global_prefetch_b8 for next K-tile
            __builtin_prefetch(x + (size_t)(gm0 + (tid >> 3)) * Dq + k0 + 32, 0, 1);
            __builtin_prefetch(wz + (size_t)(k0 + 32 + (tid >> 7)) * Dq + gn0 + (tid & 127), 0, 1);
        }
        __syncthreads();

        v16bf bf0 = load_frag_b(Bt, wn, LDP32);
        v16bf bf1 = load_frag_b(Bt, wn + 16, LDP32);
#pragma unroll
        for (int i = 0; i < 4; ++i) {
            v16bf af = load_frag_a(As, wm + i * 16, LDP32);
            acc[i][0] = wmma_bf16(af, bf0, acc[i][0]);
            acc[i][1] = wmma_bf16(af, bf1, acc[i][1]);
        }
        __syncthreads();
    }

    const int rbase = (lane >> 4) << 3;
    const int cc    = lane & 15;
#pragma unroll
    for (int i = 0; i < 4; ++i)
#pragma unroll
        for (int j = 0; j < 2; ++j) {
            int r0 = gm0 + wm + i * 16 + rbase;
            int c0 = gn0 + wn + j * 16 + cc;
#pragma unroll
            for (int vr = 0; vr < 8; ++vr)
                out[(size_t)(r0 + vr) * Dq + c0] = f2bf(acc[i][j][vr]);
        }
}

// ---------------------------------------------------------------------------
// K2: S = Q @ K^T * scale. grid(16, 16, 4), block 256. BK=64.
// TDM double-buffered 128x64 tiles; 16 WMMAs per wave per barrier pair.
// ---------------------------------------------------------------------------
__global__ __launch_bounds__(256) void scores_gemm(const __bf16* __restrict__ Qb,
                                                   const __bf16* __restrict__ Kb,
                                                   __bf16* __restrict__ Sc) {
    __shared__ __align__(16) __bf16 Qs[2][128 * LDP64];
    __shared__ __align__(16) __bf16 Ks[2][128 * LDP64];

    const int tid  = threadIdx.x;
    const int wave = tid >> 5;
    const int lane = tid & 31;
    const int gm0  = blockIdx.y * 128;   // query rows
    const int gn0  = blockIdx.x * 128;   // key rows
    const __bf16* Qt = Qb + (size_t)blockIdx.z * Sq * Dq + (size_t)gm0 * Dq;
    const __bf16* Kt = Kb + (size_t)blockIdx.z * Sq * Dq + (size_t)gn0 * Dq;
    __bf16* Sm = Sc + (size_t)blockIdx.z * Sq * Sq;

    const int wm = (wave >> 2) * 64;
    const int wn = (wave & 3) * 32;

    v8f acc[4][2] = {};

    if (wave == 0) {   // prologue: DMA first 128x64 tile pair into buffer 0
        tdm_load_tile_bf16(Qt, lds_off(&Qs[0][0]), Dq, Sq, 64, 128, Dq, 4);
        tdm_load_tile_bf16(Kt, lds_off(&Ks[0][0]), Dq, Sq, 64, 128, Dq, 4);
    }

    const int NIT = Dq / 64;             // 16
    for (int it = 0; it < NIT; ++it) {
        const int buf = it & 1;
        if (wave == 0) {
            if (it + 1 < NIT) {          // issue next tile pair, then wait current
                const int k1 = (it + 1) * 64;
                tdm_load_tile_bf16(Qt + k1, lds_off(&Qs[buf ^ 1][0]), Dq, Sq, 64, 128, Dq, 4);
                tdm_load_tile_bf16(Kt + k1, lds_off(&Ks[buf ^ 1][0]), Dq, Sq, 64, 128, Dq, 4);
                __builtin_amdgcn_s_wait_tensorcnt(2);
            } else {
                __builtin_amdgcn_s_wait_tensorcnt(0);
            }
        }
        __syncthreads();

#pragma unroll
        for (int ks = 0; ks < 2; ++ks) {  // two 32-deep WMMA steps per tile
            const __bf16* Qp = &Qs[buf][0] + ks * 32;
            const __bf16* Kp = &Ks[buf][0] + ks * 32;
            v16bf bf0 = load_frag_b(Kp, wn, LDP64);
            v16bf bf1 = load_frag_b(Kp, wn + 16, LDP64);
#pragma unroll
            for (int i = 0; i < 4; ++i) {
                v16bf af = load_frag_a(Qp, wm + i * 16, LDP64);
                acc[i][0] = wmma_bf16(af, bf0, acc[i][0]);
                acc[i][1] = wmma_bf16(af, bf1, acc[i][1]);
            }
        }
        __syncthreads();
    }

    const float scale = 0.03125f;  // 1/sqrt(1024)
    const int rbase = (lane >> 4) << 3;
    const int cc    = lane & 15;
#pragma unroll
    for (int i = 0; i < 4; ++i)
#pragma unroll
        for (int j = 0; j < 2; ++j) {
            int r0 = gm0 + wm + i * 16 + rbase;
            int c0 = gn0 + wn + j * 16 + cc;
#pragma unroll
            for (int vr = 0; vr < 8; ++vr)
                Sm[(size_t)(r0 + vr) * Sq + c0] = f2bf(acc[i][j][vr] * scale);
        }
}

// ---------------------------------------------------------------------------
// K3: softmax over each row of S (in place, bf16). grid(8192), block 256.
// ---------------------------------------------------------------------------
__global__ __launch_bounds__(256) void softmax_rows(__bf16* __restrict__ Sc) {
    __shared__ float red[8];
    const int tid = threadIdx.x;
    __bf16* p = Sc + (size_t)blockIdx.x * Sq;

    float v[8];
    float mx = -3.0e38f;
#pragma unroll
    for (int i = 0; i < 8; ++i) {
        v[i] = bf2f(p[tid + i * 256]);
        mx = fmaxf(mx, v[i]);
    }
#pragma unroll
    for (int off = 16; off > 0; off >>= 1)
        mx = fmaxf(mx, __shfl_xor(mx, off, 32));
    if ((tid & 31) == 0) red[tid >> 5] = mx;
    __syncthreads();
    mx = red[0];
#pragma unroll
    for (int i = 1; i < 8; ++i) mx = fmaxf(mx, red[i]);
    __syncthreads();

    float sum = 0.0f;
#pragma unroll
    for (int i = 0; i < 8; ++i) {
        v[i] = __expf(v[i] - mx);
        sum += v[i];
    }
#pragma unroll
    for (int off = 16; off > 0; off >>= 1)
        sum += __shfl_xor(sum, off, 32);
    if ((tid & 31) == 0) red[tid >> 5] = sum;
    __syncthreads();
    sum = 0.0f;
#pragma unroll
    for (int i = 0; i < 8; ++i) sum += red[i];

    const float inv = 1.0f / sum;
#pragma unroll
    for (int i = 0; i < 8; ++i)
        p[tid + i * 256] = f2bf(v[i] * inv);
}

// ---------------------------------------------------------------------------
// K4: out = P @ V (fp32 out). grid(8, 16, 4), block 256. BK=64.
// P tile via double-buffered TDM; V tile transposed through VGPRs.
// ---------------------------------------------------------------------------
__global__ __launch_bounds__(256) void pv_gemm(const __bf16* __restrict__ Sc,
                                               const __bf16* __restrict__ Vb,
                                               float* __restrict__ out) {
    __shared__ __align__(16) __bf16 Ps[2][128 * LDP64];
    __shared__ __align__(16) __bf16 Vt[128 * LDP64];

    const int tid  = threadIdx.x;
    const int wave = tid >> 5;
    const int lane = tid & 31;
    const int gm0  = blockIdx.y * 128;   // query rows
    const int gn0  = blockIdx.x * 128;   // output cols (e)
    const __bf16* Pt = Sc + (size_t)blockIdx.z * Sq * Sq + (size_t)gm0 * Sq;
    const __bf16* Vm = Vb + (size_t)blockIdx.z * Sq * Dq;
    float* Om = out + (size_t)blockIdx.z * Sq * Dq;

    const int wm = (wave >> 2) * 64;
    const int wn = (wave & 3) * 32;

    v8f acc[4][2] = {};

    if (wave == 0) {   // prologue: DMA first 128x64 P tile into buffer 0
        tdm_load_tile_bf16(Pt, lds_off(&Ps[0][0]), Sq, Sq, 64, 128, Sq, 4);
    }

    const int NIT = Sq / 64;             // 32
    for (int it = 0; it < NIT; ++it) {
        const int buf = it & 1;
        const int k0  = it * 64;

        // stage V tile 64x128 transposed -> Vt[e][k] (all threads)
        const unsigned short* Vus = (const unsigned short*)Vm;
#pragma unroll
        for (int p = 0; p < 8; ++p) {
            int e  = tid & 127;
            int g  = (tid >> 7) + p * 2;    // 0..15
            int kk = g << 2;                // 0..60
            const unsigned short* vp = Vus + (size_t)(k0 + kk) * Dq + gn0 + e;
            ushort4 h;
            h.x = vp[0 * Dq]; h.y = vp[1 * Dq]; h.z = vp[2 * Dq]; h.w = vp[3 * Dq];
            *(ushort4*)(&Vt[e * LDP64 + kk]) = h;
        }

        if (wave == 0) {
            if (it + 1 < NIT) {
                tdm_load_tile_bf16(Pt + (it + 1) * 64, lds_off(&Ps[buf ^ 1][0]),
                                   Sq, Sq, 64, 128, Sq, 4);
                __builtin_amdgcn_s_wait_tensorcnt(1);
            } else {
                __builtin_amdgcn_s_wait_tensorcnt(0);
            }
        }
        __syncthreads();

#pragma unroll
        for (int ks = 0; ks < 2; ++ks) {
            const __bf16* Pp = &Ps[buf][0] + ks * 32;
            const __bf16* Vp = &Vt[0] + ks * 32;
            v16bf bf0 = load_frag_b(Vp, wn, LDP64);
            v16bf bf1 = load_frag_b(Vp, wn + 16, LDP64);
#pragma unroll
            for (int i = 0; i < 4; ++i) {
                v16bf af = load_frag_a(Pp, wm + i * 16, LDP64);
                acc[i][0] = wmma_bf16(af, bf0, acc[i][0]);
                acc[i][1] = wmma_bf16(af, bf1, acc[i][1]);
            }
        }
        __syncthreads();
    }

    const int rbase = (lane >> 4) << 3;
    const int cc    = lane & 15;
#pragma unroll
    for (int i = 0; i < 4; ++i)
#pragma unroll
        for (int j = 0; j < 2; ++j) {
            int r0 = gm0 + wm + i * 16 + rbase;
            int c0 = gn0 + wn + j * 16 + cc;
#pragma unroll
            for (int vr = 0; vr < 8; ++vr)
                Om[(size_t)(r0 + vr) * Dq + c0] = acc[i][j][vr];
        }
}

// ---------------------------------------------------------------------------
extern "C" void kernel_launch(void* const* d_in, const int* in_sizes, int n_in,
                              void* d_out, int out_size, void* d_ws, size_t ws_size,
                              hipStream_t stream) {
    const float* x = (const float*)d_in[0];      // [4, 2048, 1024] fp32
    const float* w = (const float*)d_in[1];      // [3, 1024, 1024] fp32
    float* out = (float*)d_out;                  // [4, 2048, 1024] fp32

    char* ws = (char*)d_ws;
    const size_t MB = 1024ull * 1024ull;
    __bf16* Qb = (__bf16*)(ws + 0 * MB);         // 16 MB
    __bf16* Kb = (__bf16*)(ws + 16 * MB);        // 16 MB
    __bf16* Vb = (__bf16*)(ws + 32 * MB);        // 16 MB
    __bf16* Sc = (__bf16*)(ws + 48 * MB);        // 32 MB (scores/probs, in L2)

    qkv_gemm<<<dim3(8, 64, 3), 256, 0, stream>>>(x, w, Qb);
    scores_gemm<<<dim3(16, 16, 4), 256, 0, stream>>>(Qb, Kb, Sc);
    softmax_rows<<<dim3(Bq * Sq), 256, 0, stream>>>(Sc);
    pv_gemm<<<dim3(8, 16, 4), 256, 0, stream>>>(Sc, Vb, out);
}
